// DDSP_566935683241
// MI455X (gfx1250) — compile-verified, matrix operand order checked by reading, not run on hardware
//
#include <hip/hip_runtime.h>
#include <hip/hip_bf16.h>
#include <math.h>

#define HIDN 512
#define NB 4
#define NT 1000
#define NROWS (NB * NT)   // 4000
#define NHARM 64
#define UPS 160
#define NSAMP (NT * UPS)  // 160000 per batch
#define X2LD 544          // 514 padded to 17*32
#define EPS_LN 1e-5f

typedef __attribute__((ext_vector_type(16))) _Float16 v16h;
typedef __attribute__((ext_vector_type(8)))  _Float16 v8h;
typedef __attribute__((ext_vector_type(8)))  float    v8f;

union F16x16 { v16h v; v8h h[2]; };

// Load one 16x32 (f16) A/B fragment from a row-major f16 matrix.
// Row of fragment = lane&15 ; K chunks per CDNA5 layout:
// lanes 0-15: K = k..k+7 and k+16..k+23 ; lanes 16-31: +8.
__device__ __forceinline__ v16h ld_frag(const _Float16* p, int ld, int k) {
    int lane = threadIdx.x & 31;
    const _Float16* q = p + (size_t)(lane & 15) * ld + k + ((lane >> 4) << 3);
    F16x16 u;
    u.h[0] = *(const v8h*)(q);
    u.h[1] = *(const v8h*)(q + 16);
    return u.v;
}

#define WMMA16(a, b, c) \
    __builtin_amdgcn_wmma_f32_16x16x32_f16(false, (a), false, (b), (short)0, (c), false, false)

__device__ __forceinline__ float sigm(float x) { return 1.0f / (1.0f + expf(-x)); }

// ---------------------------------------------------------------------------
// fp32 -> fp16 weight repack with optional zero column padding
// ---------------------------------------------------------------------------
__global__ void w_repack_kernel(const float* __restrict__ src, _Float16* __restrict__ dst,
                                int rows, int scols, int dstride) {
    int idx = blockIdx.x * blockDim.x + threadIdx.x;
    int total = rows * dstride;
    if (idx >= total) return;
    int r = idx / dstride, c = idx - r * dstride;
    dst[idx] = (c < scols) ? (_Float16)src[r * scols + c] : (_Float16)0.0f;
}

// ---------------------------------------------------------------------------
// MLP layer 1 (in_dim == 1): y = x*W + b, LayerNorm, LeakyReLU -> f16
// one wave per row, 16 columns per lane
// ---------------------------------------------------------------------------
__global__ void __launch_bounds__(256)
mlp1_kernel(const float* __restrict__ x, const float* __restrict__ W,
            const float* __restrict__ b, const float* __restrict__ g,
            const float* __restrict__ be, _Float16* __restrict__ out,
            int ldo, int ocol) {
    int wave = threadIdx.x >> 5, lane = threadIdx.x & 31;
    int row = blockIdx.x * 8 + wave;
    if (row >= NROWS) return;
    float xv = x[row];
    float vals[16], s = 0.f, sq = 0.f;
    #pragma unroll
    for (int i = 0; i < 16; ++i) {
        int c = lane + 32 * i;
        float y = xv * W[c] + b[c];
        vals[i] = y; s += y; sq += y * y;
    }
    for (int m = 16; m >= 1; m >>= 1) { s += __shfl_xor(s, m, 32); sq += __shfl_xor(sq, m, 32); }
    float mean = s * (1.f / HIDN);
    float var  = sq * (1.f / HIDN) - mean * mean;
    float rstd = rsqrtf(var + EPS_LN);
    #pragma unroll
    for (int i = 0; i < 16; ++i) {
        int c = lane + 32 * i;
        float y = (vals[i] - mean) * rstd * g[c] + be[c];
        y = y > 0.f ? y : 0.01f * y;
        out[(size_t)row * ldo + ocol + c] = (_Float16)y;
    }
}

// ---------------------------------------------------------------------------
// WMMA GEMM: out = A(f16, MxK) @ W(f16, NxK)^T + bias, epilogue:
//   DO_LN=true : LayerNorm(512) + LeakyReLU, f16 store (N==512, grid.y==1)
//   DO_LN=false: plain bias add, f16 store
// block = 256 threads (8 waves), each block: 16 rows x 512 cols
// ---------------------------------------------------------------------------
template <bool DO_LN>
__global__ void __launch_bounds__(256)
gemm_kernel(const _Float16* __restrict__ A, int lda,
            const _Float16* __restrict__ Wm, int ldw,
            const float* __restrict__ bias,
            const float* __restrict__ gamma, const float* __restrict__ beta,
            _Float16* __restrict__ out, int ldo, int ocol, int K) {
    int wave = threadIdx.x >> 5, lane = threadIdx.x & 31;
    int row0  = blockIdx.x * 16;
    int ncol0 = blockIdx.y * 512;
    int nbase = ncol0 + wave * 64;
    const _Float16* Arow = A + (size_t)row0 * lda;

    v8f acc[4] = {};
    for (int k = 0; k < K; k += 32) {
        v16h aF = ld_frag(Arow, lda, k);
        #pragma unroll
        for (int j = 0; j < 4; ++j) {
            v16h bF = ld_frag(Wm + (size_t)(nbase + j * 16) * ldw, ldw, k);
            acc[j] = WMMA16(aF, bF, acc[j]);
        }
    }

    int nloc = lane & 15, mhi = (lane >> 4) * 8;
    if constexpr (DO_LN) {
        __shared__ float sm[16][512];
        #pragma unroll
        for (int j = 0; j < 4; ++j) {
            int ccol = wave * 64 + j * 16 + nloc;
            float bv = bias[ccol];
            #pragma unroll
            for (int v = 0; v < 8; ++v) sm[v + mhi][ccol] = acc[j][v] + bv;
        }
        __syncthreads();
        for (int rr = wave * 2; rr < wave * 2 + 2; ++rr) {
            float s = 0.f, sq = 0.f;
            #pragma unroll
            for (int i = 0; i < 16; ++i) { float y = sm[rr][lane + 32 * i]; s += y; sq += y * y; }
            for (int m = 16; m >= 1; m >>= 1) { s += __shfl_xor(s, m, 32); sq += __shfl_xor(sq, m, 32); }
            float mean = s * (1.f / HIDN);
            float var  = sq * (1.f / HIDN) - mean * mean;
            float rstd = rsqrtf(var + EPS_LN);
            #pragma unroll
            for (int i = 0; i < 16; ++i) {
                int c = lane + 32 * i;
                float y = (sm[rr][c] - mean) * rstd * gamma[c] + beta[c];
                y = y > 0.f ? y : 0.01f * y;
                out[(size_t)(row0 + rr) * ldo + ocol + c] = (_Float16)y;
            }
        }
    } else {
        #pragma unroll
        for (int j = 0; j < 4; ++j) {
            int n = nbase + j * 16 + nloc;
            float bv = bias[n];
            #pragma unroll
            for (int v = 0; v < 8; ++v) {
                int m = v + mhi;
                out[(size_t)(row0 + m) * ldo + ocol + n] = (_Float16)(acc[j][v] + bv);
            }
        }
    }
}

// ---------------------------------------------------------------------------
// Fill concat tail of X2: col 512 = f0, col 513 = loudness, 514..543 = 0
// ---------------------------------------------------------------------------
__global__ void x2_tail_kernel(const float* __restrict__ f0, const float* __restrict__ ld,
                               _Float16* __restrict__ X2) {
    int idx = blockIdx.x * blockDim.x + threadIdx.x;
    if (idx >= NROWS * 32) return;
    int row = idx >> 5, c = idx & 31;
    float v = (c == 0) ? f0[row] : (c == 1) ? ld[row] : 0.0f;
    X2[(size_t)row * X2LD + 512 + c] = (_Float16)v;
}

// ---------------------------------------------------------------------------
// GRU recurrence. Single workgroup, 16 waves. h kept in LDS as 16x512 f16
// (rows 0..3 = batch, rows 4..15 stay zero-padded). Each wave owns 32 h
// columns -> 2 WMMA n-tiles in each of the r/z/n gate blocks of Whh.
// gi (= X @ Wih^T + bih) was precomputed by the big parallel GEMM.
// Writes h_t directly into X2 columns 0..511.
//
// An SGPR *offset* (always zero, but opaque to the compiler) is laundered
// through a no-op asm every timestep so LICM cannot hoist the (t-invariant)
// Whh fragment loads out of the 1000-step loop (round-1 showed it hoisting
// 768 VGPRs of fragments and spilling them all to scratch). Laundering the
// offset instead of the pointer keeps the global address space inferred on
// Whh, so these stay global_load_b128 (LOADcnt only) instead of
// flat_load_b128 (LOADcnt+DScnt, which coupled the weight stream to the LDS
// h-fragment stream in round-2 codegen).
// ---------------------------------------------------------------------------
__global__ void __launch_bounds__(512)
gru_kernel(const _Float16* __restrict__ gi,     // 4000 x 1536 f16
           const _Float16* __restrict__ Whh,    // 1536 x 512 f16
           const float* __restrict__ bhh,       // 1536
           _Float16* __restrict__ X2) {         // 4000 x 544 f16
    __shared__ _Float16 hsh[16 * 512];
    int tid = threadIdx.x;
    int wave = tid >> 5, lane = tid & 31;
    for (int i = tid; i < 16 * 512; i += 512) hsh[i] = (_Float16)0.0f;
    __syncthreads();

    int jbase = wave * 32;
    int nloc = lane & 15, mhi = (lane >> 4) * 8;

    for (int t = 0; t < NT; ++t) {
        // opaque zero offset (defeats LICM, preserves global addrspace)
        unsigned int zoff = 0;
        asm volatile("" : "+s"(zoff));
        const _Float16* Wcur = Whh + zoff;

        v8f acc[6] = {};
        #pragma unroll 4
        for (int k = 0; k < 512; k += 32) {
            v16h aF = ld_frag(hsh, 512, k);
            #pragma unroll
            for (int g = 0; g < 3; ++g) {
                #pragma unroll
                for (int i = 0; i < 2; ++i) {
                    int nrow = g * 512 + jbase + i * 16;
                    v16h bF = ld_frag(Wcur + (size_t)nrow * 512, 512, k);
                    acc[g * 2 + i] = WMMA16(aF, bF, acc[g * 2 + i]);
                }
            }
        }

        float hnew[2][8];
        #pragma unroll
        for (int i = 0; i < 2; ++i) {
            int j = jbase + i * 16 + nloc;
            float br = bhh[j], bz = bhh[512 + j], bn = bhh[1024 + j];
            #pragma unroll
            for (int v = 0; v < 8; ++v) {
                int m = v + mhi;
                float gir = 0.f, giz = 0.f, gin = 0.f, hold = 0.f;
                if (m < NB) {
                    size_t rrow = (size_t)(m * NT + t) * 1536;
                    gir  = (float)gi[rrow + j];
                    giz  = (float)gi[rrow + 512 + j];
                    gin  = (float)gi[rrow + 1024 + j];
                    hold = (float)hsh[m * 512 + j];
                }
                float r = sigm(gir + acc[0 + i][v] + br);
                float z = sigm(giz + acc[2 + i][v] + bz);
                float n = tanhf(gin + r * (acc[4 + i][v] + bn));
                hnew[i][v] = (1.f - z) * n + z * hold;
            }
        }
        __syncthreads();  // all reads of old h done
        #pragma unroll
        for (int i = 0; i < 2; ++i) {
            int j = jbase + i * 16 + nloc;
            #pragma unroll
            for (int v = 0; v < 8; ++v) {
                int m = v + mhi;
                if (m < NB) {
                    _Float16 hv = (_Float16)hnew[i][v];
                    hsh[m * 512 + j] = hv;
                    X2[(size_t)(m * NT + t) * X2LD + j] = hv;
                }
            }
        }
        __syncthreads();  // h fully updated before next step
    }
}

// ---------------------------------------------------------------------------
// Harmonic head: ctrl = scale_fn(Y @ harm_w^T + harm_b), 65 outputs/row
// ---------------------------------------------------------------------------
__global__ void harm_proj_kernel(const _Float16* __restrict__ Y,
                                 const float* __restrict__ hw, const float* __restrict__ hb,
                                 float* __restrict__ ctrl) {
    int idx = blockIdx.x * blockDim.x + threadIdx.x;
    if (idx >= NROWS * (NHARM + 1)) return;
    int row = idx / (NHARM + 1);
    int o   = idx - row * (NHARM + 1);
    const _Float16* yr = Y + (size_t)row * 512;
    const float* wr = hw + (size_t)o * 512;
    float acc = hb[o];
    for (int i = 0; i < 512; ++i) acc += (float)yr[i] * wr[i];
    float s = sigm(acc);
    ctrl[idx] = 2.0f * powf(s, 2.30258509299f) + 1e-7f;
}

// ---------------------------------------------------------------------------
// Upsample helpers + hierarchical cumsum + sine-bank synthesis
// ---------------------------------------------------------------------------
__device__ __forceinline__ float f0_interp(const float* __restrict__ f0, int b, int s) {
    float u = (s + 0.5f) * (1.0f / UPS) - 0.5f;
    float fu = floorf(u);
    float w = u - fu;
    int i0 = (int)fu, i1 = i0 + 1;
    i0 = min(max(i0, 0), NT - 1);
    i1 = min(max(i1, 0), NT - 1);
    const float* p = f0 + b * NT;
    return p[i0] * (1.f - w) + p[i1] * w;
}

__global__ void chunk_sum_kernel(const float* __restrict__ f0, float* __restrict__ csum) {
    int idx = blockIdx.x * blockDim.x + threadIdx.x;
    if (idx >= NB * NT) return;
    int b = idx / NT, c = idx - b * NT;
    float s = 0.f;
    for (int j = 0; j < UPS; ++j) s += f0_interp(f0, b, c * UPS + j);
    csum[idx] = s;
}

__global__ void prefix_kernel(const float* __restrict__ csum, float* __restrict__ cpref) {
    int b = threadIdx.x;
    if (b >= NB) return;
    float run = 0.f;
    for (int c = 0; c < NT; ++c) { cpref[b * NT + c] = run; run += csum[b * NT + c]; }
}

__global__ void __launch_bounds__(256)
synth_kernel(const float* __restrict__ f0, const float* __restrict__ cpref,
             const float* __restrict__ ctrl, float* __restrict__ out) {
    int idx = blockIdx.x * blockDim.x + threadIdx.x;
    if (idx >= NB * NSAMP) return;
    int b = idx / NSAMP;
    int s = idx - b * NSAMP;
    int c = s / UPS;

    // inclusive cumsum of f0_rs up to s  (chunk prefix + local scan)
    float acc = cpref[b * NT + c];
    for (int s2 = c * UPS; s2 <= s; ++s2) acc += f0_interp(f0, b, s2);
    float omega = acc * (2.0f * 3.14159265358979f / 16000.0f);

    // linear-resampled control frame
    float u = (s + 0.5f) * (1.0f / UPS) - 0.5f;
    float fu = floorf(u);
    float w = u - fu;
    int i0 = (int)fu, i1 = i0 + 1;
    i0 = min(max(i0, 0), NT - 1);
    i1 = min(max(i1, 0), NT - 1);
    const float* c0 = ctrl + (size_t)(b * NT + i0) * (NHARM + 1);
    const float* c1 = ctrl + (size_t)(b * NT + i1) * (NHARM + 1);

    float amp = c0[0] + (c1[0] - c0[0]) * w;
    float a[NHARM];
    float l1 = 0.f;
    #pragma unroll 8
    for (int k = 0; k < NHARM; ++k) {
        float v = c0[k + 1] + (c1[k + 1] - c0[k + 1]) * w;
        a[k] = v;
        l1 += fabsf(v);
    }
    float inv = 1.0f / fmaxf(l1, 1e-12f);
    float sum = 0.f;
    #pragma unroll 8
    for (int k = 0; k < NHARM; ++k) sum += sinf(omega * (float)(k + 1)) * a[k];
    out[idx] = amp * inv * sum;
}

// ---------------------------------------------------------------------------
// Launch
// ---------------------------------------------------------------------------
extern "C" void kernel_launch(void* const* d_in, const int* in_sizes, int n_in,
                              void* d_out, int out_size, void* d_ws, size_t ws_size,
                              hipStream_t stream) {
    // inputs in setup_inputs() dict order (pytrees flattened in order)
    const float* f0  = (const float*)d_in[0];   // 4000
    const float* ldn = (const float*)d_in[1];   // 4000
    // mlp_f0_params: [W,b,g,beta] x3  -> idx 2..13
    // mlp_ld_params: idx 14..25 ; gru: 26..29 ; mlp_out: 30..41 ; harm: 42,43
    const float* f0W1 = (const float*)d_in[2],  *f0b1 = (const float*)d_in[3];
    const float* f0g1 = (const float*)d_in[4],  *f0B1 = (const float*)d_in[5];
    const float* f0W2 = (const float*)d_in[6],  *f0b2 = (const float*)d_in[7];
    const float* f0g2 = (const float*)d_in[8],  *f0B2 = (const float*)d_in[9];
    const float* f0W3 = (const float*)d_in[10], *f0b3 = (const float*)d_in[11];
    const float* f0g3 = (const float*)d_in[12], *f0B3 = (const float*)d_in[13];
    const float* ldW1 = (const float*)d_in[14], *ldb1 = (const float*)d_in[15];
    const float* ldg1 = (const float*)d_in[16], *ldB1 = (const float*)d_in[17];
    const float* ldW2 = (const float*)d_in[18], *ldb2 = (const float*)d_in[19];
    const float* ldg2 = (const float*)d_in[20], *ldB2 = (const float*)d_in[21];
    const float* ldW3 = (const float*)d_in[22], *ldb3 = (const float*)d_in[23];
    const float* ldg3 = (const float*)d_in[24], *ldB3 = (const float*)d_in[25];
    const float* Wih  = (const float*)d_in[26]; // 1536x1024
    const float* Whh  = (const float*)d_in[27]; // 1536x512
    const float* bih  = (const float*)d_in[28];
    const float* bhh  = (const float*)d_in[29];
    const float* oW1 = (const float*)d_in[30], *ob1 = (const float*)d_in[31];
    const float* og1 = (const float*)d_in[32], *oB1 = (const float*)d_in[33];
    const float* oW2 = (const float*)d_in[34], *ob2 = (const float*)d_in[35];
    const float* og2 = (const float*)d_in[36], *oB2 = (const float*)d_in[37];
    const float* oW3 = (const float*)d_in[38], *ob3 = (const float*)d_in[39];
    const float* og3 = (const float*)d_in[40], *oB3 = (const float*)d_in[41];
    const float* harm_w = (const float*)d_in[42]; // 65x512
    const float* harm_b = (const float*)d_in[43]; // 65
    float* out = (float*)d_out;

    // workspace carve-out
    char* w = (char*)d_ws;
    auto alloc = [&](size_t bytes) { void* p = (void*)w; w += (bytes + 255) & ~(size_t)255; return p; };
    _Float16* X    = (_Float16*)alloc((size_t)NROWS * 1024 * 2); // GRU input concat
    _Float16* GI   = (_Float16*)alloc((size_t)NROWS * 1536 * 2); // X @ Wih^T + bih
    _Float16* X2   = (_Float16*)alloc((size_t)NROWS * X2LD * 2); // [h, f0, ld, pad]
    _Float16* Y1   = (_Float16*)alloc((size_t)NROWS * 512 * 2);  // ping
    _Float16* Y2   = (_Float16*)alloc((size_t)NROWS * 512 * 2);  // pong
    float*    ctrl = (float*)alloc((size_t)NROWS * (NHARM + 1) * 4);
    float*    csum = (float*)alloc((size_t)NB * NT * 4);
    float*    cprf = (float*)alloc((size_t)NB * NT * 4);
    _Float16* f0W2h = (_Float16*)alloc((size_t)512 * 512 * 2);
    _Float16* f0W3h = (_Float16*)alloc((size_t)512 * 512 * 2);
    _Float16* ldW2h = (_Float16*)alloc((size_t)512 * 512 * 2);
    _Float16* ldW3h = (_Float16*)alloc((size_t)512 * 512 * 2);
    _Float16* Wih_h = (_Float16*)alloc((size_t)1536 * 1024 * 2);
    _Float16* Whh_h = (_Float16*)alloc((size_t)1536 * 512 * 2);
    _Float16* oW1h  = (_Float16*)alloc((size_t)512 * X2LD * 2); // 514 -> 544 zero pad
    _Float16* oW2h  = (_Float16*)alloc((size_t)512 * 512 * 2);
    _Float16* oW3h  = (_Float16*)alloc((size_t)512 * 512 * 2);

    auto rep = [&](const float* s, _Float16* d, int r, int sc, int ds) {
        int total = r * ds;
        w_repack_kernel<<<(total + 255) / 256, 256, 0, stream>>>(s, d, r, sc, ds);
    };
    rep(f0W2, f0W2h, 512, 512, 512);
    rep(f0W3, f0W3h, 512, 512, 512);
    rep(ldW2, ldW2h, 512, 512, 512);
    rep(ldW3, ldW3h, 512, 512, 512);
    rep(Wih,  Wih_h, 1536, 1024, 1024);
    rep(Whh,  Whh_h, 1536, 512, 512);
    rep(oW1,  oW1h,  512, 514, X2LD);
    rep(oW2,  oW2h,  512, 512, 512);
    rep(oW3,  oW3h,  512, 512, 512);

    dim3 blk(256);
    dim3 g16(NROWS / 16, 1);

    // f0 MLP: L1 -> Y1 ; L2 -> Y2 ; L3 -> X[:, 0:512]
    mlp1_kernel<<<NROWS / 8, blk, 0, stream>>>(f0, f0W1, f0b1, f0g1, f0B1, Y1, 512, 0);
    gemm_kernel<true><<<g16, blk, 0, stream>>>(Y1, 512, f0W2h, 512, f0b2, f0g2, f0B2, Y2, 512, 0, 512);
    gemm_kernel<true><<<g16, blk, 0, stream>>>(Y2, 512, f0W3h, 512, f0b3, f0g3, f0B3, X, 1024, 0, 512);
    // loudness MLP -> X[:, 512:1024]
    mlp1_kernel<<<NROWS / 8, blk, 0, stream>>>(ldn, ldW1, ldb1, ldg1, ldB1, Y1, 512, 0);
    gemm_kernel<true><<<g16, blk, 0, stream>>>(Y1, 512, ldW2h, 512, ldb2, ldg2, ldB2, Y2, 512, 0, 512);
    gemm_kernel<true><<<g16, blk, 0, stream>>>(Y2, 512, ldW3h, 512, ldb3, ldg3, ldB3, X, 1024, 512, 512);

    // gi = X @ Wih^T + bih  (parallel part of the GRU), N=1536 -> grid.y=3
    gemm_kernel<false><<<dim3(NROWS / 16, 3), blk, 0, stream>>>(X, 1024, Wih_h, 1024, bih,
                                                                nullptr, nullptr, GI, 1536, 0, 1024);
    // concat tail of X2 (f0, loudness, zero pad)
    x2_tail_kernel<<<(NROWS * 32 + 255) / 256, blk, 0, stream>>>(f0, ldn, X2);
    // sequential GRU recurrence -> X2[:, 0:511]
    gru_kernel<<<1, 512, 0, stream>>>(GI, Whh_h, bhh, X2);

    // output MLP: (X2 544) -> Y1 -> Y2 -> Y1
    gemm_kernel<true><<<g16, blk, 0, stream>>>(X2, X2LD, oW1h, X2LD, ob1, og1, oB1, Y1, 512, 0, X2LD);
    gemm_kernel<true><<<g16, blk, 0, stream>>>(Y1, 512, oW2h, 512, ob2, og2, oB2, Y2, 512, 0, 512);
    gemm_kernel<true><<<g16, blk, 0, stream>>>(Y2, 512, oW3h, 512, ob3, og3, oB3, Y1, 512, 0, 512);

    // harmonic head + scale_fn
    harm_proj_kernel<<<(NROWS * (NHARM + 1) + 255) / 256, blk, 0, stream>>>(Y1, harm_w, harm_b, ctrl);

    // phase cumsum (hierarchical) + sine-bank synthesis
    chunk_sum_kernel<<<(NB * NT + 255) / 256, blk, 0, stream>>>(f0, csum);
    prefix_kernel<<<1, 32, 0, stream>>>(csum, cprf);
    synth_kernel<<<(NB * NSAMP + 255) / 256, blk, 0, stream>>>(f0, cprf, ctrl, out);
}
